// All_att_85529978733134
// MI455X (gfx1250) — matrix operand order repaired and checked by reading, not compiled, not run on hardware
//
#include <hip/hip_runtime.h>
#include <stdint.h>

// ---------------------------------------------------------------------------
// Problem constants (reference: B=32, C=64, H=W=32, N=1024)
// ---------------------------------------------------------------------------
#define BATCH 32
#define CCH   64
#define NSP   1024

typedef __attribute__((ext_vector_type(16))) __bf16 v16bf;
typedef __attribute__((ext_vector_type(8)))  float  v8f;
typedef unsigned int v4u  __attribute__((ext_vector_type(4)));
typedef int          v8i_t __attribute__((ext_vector_type(8)));
typedef int          v4i_t __attribute__((ext_vector_type(4)));

#if __has_builtin(__builtin_amdgcn_tensor_load_to_lds) && \
    __has_builtin(__builtin_amdgcn_s_wait_tensorcnt)
#define HAVE_TDM 1
#else
#define HAVE_TDM 0
#endif

__device__ __forceinline__ __bf16 f32_to_bf16(float f) {
    union { float f; uint32_t u; } v; v.f = f;
    uint32_t r = v.u + 0x7FFFu + ((v.u >> 16) & 1u);   // round-to-nearest-even
    union { uint16_t u; __bf16 b; } o; o.u = (uint16_t)(r >> 16);
    return o.b;
}

// B operands are stored K-panel packed: Bp[k/32][n][k%32]  (bf16)
//   -> a WMMA B fragment (one column, 16 consecutive K) is 32 contiguous bytes.
__device__ __forceinline__ size_t bpack_idx(int k, int n, int npar) {
    return ((size_t)(k >> 5) * npar + n) * 32 + (k & 31);
}

#if HAVE_TDM
// ---------------------------------------------------------------------------
// Tensor Data Mover: 2-D tile DMA global -> LDS (CDNA5 ISA ch.8 D# layout).
//   rows x row_elems bf16 tile; global row stride = src_stride (elements);
//   LDS rows padded with (pad_amount_code+1) dwords every 2^(pad_code+1) dwords.
// Issued by ONE wave (uniform operands); EXEC ignored by TDM.
// 6-arg builtin form (clang-23 / amdgpu-toolchain).
// ---------------------------------------------------------------------------
__device__ __forceinline__ void tdm_load_tile_bf16(
    const void* gaddr, uint32_t lds_byte_off,
    uint32_t rows, uint32_t row_elems, uint32_t src_stride,
    uint32_t pad_interval_code, uint32_t pad_amount_code)
{
    const uint64_t ga = (uint64_t)(uintptr_t)gaddr;
    v4u g0;
    g0[0] = 1u;                                   // count=1, user descriptor
    g0[1] = lds_byte_off;                         // lds_addr
    g0[2] = (uint32_t)ga;                         // global_addr[31:0]
    g0[3] = ((uint32_t)(ga >> 32) & 0x01FFFFFFu)  // global_addr[56:32]
          | (2u << 30);                           // type=2 ("image")
    const uint32_t d0 = src_stride;               // tensor_dim0 (OOB extent)
    const uint32_t d1 = rows;                     // tensor_dim1
    v8i_t g1;
    g1[0] = (int)((1u << 16)                      // data_size=1 -> 2 bytes
          | (1u << 20)                            // pad_enable
          | (pad_interval_code << 22)             // pad every 2^(code+1) dwords
          | (pad_amount_code << 25));             // pad (code+1) dwords
    g1[1] = (int)((d0 & 0xFFFFu) << 16);          // tensor_dim0[15:0]  @ bit48
    g1[2] = (int)((d0 >> 16) | ((d1 & 0xFFFFu) << 16));
    g1[3] = (int)((d1 >> 16) | (row_elems << 16));// tile_dim0 @ bit112
    g1[4] = (int)(rows & 0xFFFFu);                // tile_dim1; tile_dim2=0
    g1[5] = (int)src_stride;                      // tensor_dim0_stride[31:0]
    g1[6] = 0;                                    // stride hi / dim1_stride
    g1[7] = 0;
    const v4i_t z4 = {0, 0, 0, 0};                // groups 2/3 unused (2-D)
    const v8i_t z8 = {0, 0, 0, 0, 0, 0, 0, 0};    // extra group (6-arg form)
    __builtin_amdgcn_tensor_load_to_lds(g0, g1, z4, z4, z8, 0);
}

__device__ __forceinline__ uint32_t lds_offset_of(const void* p) {
    return (uint32_t)(uintptr_t)(__attribute__((address_space(3))) const void*)p;
}
#endif

// ---------------------------------------------------------------------------
// Kernel 1: w_mk (f32 [N,N]) -> bf16, transposed AND K-panel packed:
//   wmkT_p[bpack_idx(m, k, N)] = w_mk[k][m]
// ---------------------------------------------------------------------------
__global__ __launch_bounds__(256)
void wmk_transpose_kernel(const float* __restrict__ w_mk, __bf16* __restrict__ wmkTp) {
    __shared__ float tile[32][33];
    const int k0 = blockIdx.y * 32, m0 = blockIdx.x * 32;   // m0 multiple of 32
    const int tx = threadIdx.x & 31, ty = threadIdx.x >> 5; // 8 row-groups
    for (int r = ty; r < 32; r += 8)
        tile[r][tx] = w_mk[(size_t)(k0 + r) * NSP + m0 + tx];
    __syncthreads();
    for (int r = ty; r < 32; r += 8)   // element (m = m0+r, k = k0+tx)
        wmkTp[((size_t)(m0 >> 5) * NSP + (k0 + tx)) * 32 + r]
            = f32_to_bf16(tile[tx][r]);
}

// ---------------------------------------------------------------------------
// Kernel 2: fused channel projections (weights are wave-uniform -> s_loads).
//   phi  (B operand of GEMM1): phi_p [b][o/32][n][o%32]      (bf16)
//   theta (A operand of GEMM1): theta[b][n][o] row-major      (bf16)
//   gmv  (B operand of GEMM3): gmv_p [b][n/32][d][n%32]      (bf16)
// ---------------------------------------------------------------------------
__global__ __launch_bounds__(128)
void proj_kernel(const float* __restrict__ x,
                 const float* __restrict__ w_phi,
                 const float* __restrict__ w_theta,
                 const float* __restrict__ w_g,
                 const float* __restrict__ w_mv,
                 __bf16* __restrict__ phi_p,
                 __bf16* __restrict__ theta,
                 __bf16* __restrict__ gmv_p) {
    __shared__ float s_x[CCH][128];
    __shared__ float s_g[CCH][128];
    const int b  = blockIdx.y;
    const int n0 = blockIdx.x * 128;
    const int tid = threadIdx.x;
    const int n  = n0 + tid;

    const float* xb = x + (size_t)b * CCH * NSP;
    for (int c = 0; c < CCH; ++c)
        s_x[c][tid] = xb[(size_t)c * NSP + n];
    __syncthreads();

    __bf16* phi_b = phi_p + (size_t)b * CCH * NSP;
    for (int o = 0; o < CCH; ++o) {
        float ap = 0.f, at = 0.f, ag = 0.f;
        #pragma unroll 8
        for (int c = 0; c < CCH; ++c) {
            const float xv = s_x[c][tid];
            ap = fmaf(w_phi  [o * CCH + c], xv, ap);
            at = fmaf(w_theta[o * CCH + c], xv, at);
            ag = fmaf(w_g    [o * CCH + c], xv, ag);
        }
        phi_b[bpack_idx(o, n, NSP)] = f32_to_bf16(ap);       // k=o, col=n
        theta[((size_t)b * NSP + n) * CCH + o] = f32_to_bf16(at);
        s_g[o][tid] = ag;
    }
    __syncthreads();
    __bf16* gmv_b = gmv_p + (size_t)b * NSP * CCH;
    for (int d = 0; d < CCH; ++d) {
        float a = 0.f;
        #pragma unroll 8
        for (int c = 0; c < CCH; ++c)
            a = fmaf(w_mv[d * CCH + c], s_g[c][tid], a);
        gmv_b[bpack_idx(n, d, CCH)] = f32_to_bf16(a);        // k=n(spatial), col=d
    }
}

// ---------------------------------------------------------------------------
// Kernel 3: batched bf16 GEMM with v_wmma_f32_16x16x32_bf16.
//   C[b] = A[b] (M x K, row-major lda) * Bp[b] (K x N, K-panel packed)
// Block: 256 threads = 8 waves; block tile 128(M) x 64(N); wave tile 32x32
// (2x2 WMMA tiles); K-step 32. Tiles arrive in LDS via the Tensor Data Mover
// (descriptor issued by wave 0, TENSORcnt-tracked); TDM pad fields create the
// bank-spread strides (A: 34 halves/row, B: 40 halves/column-row = 80 B,
// keeping B fragments 16B-aligned for ds_load_b128).
// ---------------------------------------------------------------------------
template <bool OUT_BF16>
__global__ __launch_bounds__(256)
void gemm_bf16_kernel(const __bf16* __restrict__ A,
                      const __bf16* __restrict__ Bp,
                      void* __restrict__ Cout,
                      int M, int Npar, int K,
                      int lda, int ldc,
                      long long strideA, long long strideB, long long strideC) {
    constexpr int LDA_SH = 34;   // halves: 32 + 1 pad dword (16 dwords + 1)
    constexpr int LDB_SH = 40;   // halves: 32 + 4 pad dwords -> 80 B, 16B-aligned
    __shared__ __bf16 sA[128 * LDA_SH];
    __shared__ __bf16 sB[64  * LDB_SH];

    const int b = blockIdx.z;
    A  += (size_t)b * strideA;
    Bp += (size_t)b * strideB;

    const int tid   = threadIdx.x;
    const int lane  = tid & 31;
    const int wave  = tid >> 5;
    const int waveM = wave >> 1;          // 0..3
    const int waveN = wave & 1;           // 0..1
    const int m0    = blockIdx.y * 128;
    const int n0    = blockIdx.x * 64;

    const int lhalf = (lane < 16) ? 0 : 1;  // which half-wave
    const int lrow  = lane & 15;            // A row / B col / D col index

    v8f acc[2][2] = {};

    for (int k0 = 0; k0 < K; k0 += 32) {
        // packed B panel for this K-step: 64 cols x 32 halves, contiguous
        const __bf16* Bpanel = Bp + ((size_t)(k0 >> 5) * Npar + n0) * 32;
        __syncthreads();
#if HAVE_TDM
        if (wave == 0) {
            // A tile: 128 rows x 32 halves; +1 pad dword / 16 dwords -> stride 34
            tdm_load_tile_bf16(A + (size_t)m0 * lda + k0, lds_offset_of(sA),
                               128u, 32u, (uint32_t)lda, 3u, 0u);
            // B panel: 64 "rows" (columns) x 32 halves; +4 pad dwords -> stride 40
            tdm_load_tile_bf16(Bpanel, lds_offset_of(sB),
                               64u, 32u, 32u, 3u, 3u);
            __builtin_amdgcn_s_wait_tensorcnt(0);
        }
#else
        // Fallback: cooperative VGPR-staged copy.
        #pragma unroll
        for (int i = tid; i < 512; i += 256) {
            const int r  = i >> 2;
            const int cc = (i & 3) * 8;
            const uint4 v = *(const uint4*)(A + (size_t)(m0 + r) * lda + k0 + cc);
            uint32_t* d = (uint32_t*)&sA[r * LDA_SH + cc];
            d[0] = v.x; d[1] = v.y; d[2] = v.z; d[3] = v.w;
        }
        {
            const int r  = tid >> 2;          // 64 cols x 4 uint4 each
            const int cc = (tid & 3) * 8;
            const uint4 v = *(const uint4*)(Bpanel + (size_t)r * 32 + cc);
            uint32_t* d = (uint32_t*)&sB[r * LDB_SH + cc];
            d[0] = v.x; d[1] = v.y; d[2] = v.z; d[3] = v.w;
        }
#endif
        if (k0 + 32 < K) {  // hint next K tile -> global_prefetch_b8
            __builtin_prefetch(A + (size_t)(m0 + (tid >> 1)) * lda + k0 + 32, 0, 1);
            __builtin_prefetch(Bp + ((size_t)((k0 >> 5) + 1) * Npar + n0) * 32
                                  + (size_t)tid * 8, 0, 1);
        }
        __syncthreads();

        // ---- build A fragments (ISA 7.12.2: 16-bit A 16x32) ----
        v16bf afrag[2];
        #pragma unroll
        for (int tm = 0; tm < 2; ++tm) {
            union { v16bf v; uint32_t u[8]; } fa;
            const int row = waveM * 32 + tm * 16 + lrow;
            #pragma unroll
            for (int i = 0; i < 8; ++i) {
                const int kk = lhalf * 16 + (i << 1);
                fa.u[i] = *(const uint32_t*)&sA[row * LDA_SH + kk];
            }
            afrag[tm] = fa.v;
        }
        // ---- build B fragments: 32 contiguous bytes per lane (packed) ----
        v16bf bfrag[2];
        #pragma unroll
        for (int tn = 0; tn < 2; ++tn) {
            union { v16bf v; uint32_t u[8]; } fb;
            const int col = waveN * 32 + tn * 16 + lrow;
            const uint32_t* bp = (const uint32_t*)&sB[col * LDB_SH + lhalf * 16];
            #pragma unroll
            for (int i = 0; i < 8; ++i) fb.u[i] = bp[i];
            bfrag[tn] = fb.v;
        }
        // ---- 2x2 WMMA tiles ----
        #pragma unroll
        for (int tm = 0; tm < 2; ++tm)
            #pragma unroll
            for (int tn = 0; tn < 2; ++tn)
                acc[tm][tn] = __builtin_amdgcn_wmma_f32_16x16x32_bf16(
                    false, afrag[tm], false, bfrag[tn],
                    (short)0, acc[tm][tn], false, false);
    }

    // ---- store D (ISA 7.12.2: 32-bit C/D 16x16) ----
    #pragma unroll
    for (int tm = 0; tm < 2; ++tm) {
        #pragma unroll
        for (int tn = 0; tn < 2; ++tn) {
            const int mb = m0 + waveM * 32 + tm * 16 + lhalf * 8;
            const int nn = n0 + waveN * 32 + tn * 16 + lrow;
            #pragma unroll
            for (int i = 0; i < 8; ++i) {
                const size_t idx = (size_t)b * strideC + (size_t)(mb + i) * ldc + nn;
                if (OUT_BF16) ((__bf16*)Cout)[idx] = f32_to_bf16(acc[tm][tn][i]);
                else          ((float*) Cout)[idx] = acc[tm][tn][i];
            }
        }
    }
}

// ---------------------------------------------------------------------------
// Kernel 4: softmax over axis=1 (column-wise over n for each (b,k) column).
// att2 f32 [b][n][k] -> att3 bf16 (same layout). Coalesced across k.
// ---------------------------------------------------------------------------
__global__ __launch_bounds__(256)
void softmax_cols_kernel(const float* __restrict__ att2, __bf16* __restrict__ att3) {
    const int b = blockIdx.y;
    const int k = blockIdx.x * 256 + threadIdx.x;
    const float* p = att2 + (size_t)b * NSP * NSP + k;
    float mx = -3.4e38f;
    for (int n = 0; n < NSP; ++n) mx = fmaxf(mx, p[(size_t)n * NSP]);
    float s = 0.f;
    for (int n = 0; n < NSP; ++n) s += __expf(p[(size_t)n * NSP] - mx);
    const float inv = 1.f / s;
    __bf16* q = att3 + (size_t)b * NSP * NSP + k;
    for (int n = 0; n < NSP; ++n)
        q[(size_t)n * NSP] = f32_to_bf16(__expf(p[(size_t)n * NSP] - mx) * inv);
}

// ---------------------------------------------------------------------------
// Kernel 5: final mask conv + residual:
//   y[b][o][n] = x[b][o][n] + sum_c w_mask[o][c] * out[b][n][c]
// ---------------------------------------------------------------------------
__global__ __launch_bounds__(128)
void mask_add_kernel(const float* __restrict__ outbnc,
                     const float* __restrict__ w_mask,
                     const float* __restrict__ x,
                     float* __restrict__ y) {
    __shared__ float s_o[CCH][128];
    const int b  = blockIdx.y;
    const int n0 = blockIdx.x * 128;
    const int tid = threadIdx.x;
    const float* src = outbnc + ((size_t)b * NSP + n0) * CCH;
    for (int i = tid; i < 128 * CCH; i += 128) {
        const int nn = i >> 6, c = i & 63;
        s_o[c][nn] = src[i];
    }
    __syncthreads();
    const int n = n0 + tid;
    for (int o = 0; o < CCH; ++o) {
        float acc = 0.f;
        #pragma unroll 8
        for (int c = 0; c < CCH; ++c)
            acc = fmaf(w_mask[o * CCH + c], s_o[c][tid], acc);
        const size_t idx = ((size_t)b * CCH + o) * NSP + n;
        y[idx] = acc + x[idx];
    }
}

// ---------------------------------------------------------------------------
// Launcher
// ---------------------------------------------------------------------------
extern "C" void kernel_launch(void* const* d_in, const int* in_sizes, int n_in,
                              void* d_out, int out_size, void* d_ws, size_t ws_size,
                              hipStream_t stream) {
    const float* x       = (const float*)d_in[0];
    const float* w_phi   = (const float*)d_in[1];
    const float* w_theta = (const float*)d_in[2];
    const float* w_g     = (const float*)d_in[3];
    const float* w_mask  = (const float*)d_in[4];
    const float* w_mv    = (const float*)d_in[5];
    const float* w_mk    = (const float*)d_in[6];
    float* y = (float*)d_out;

    // Workspace carve-up (elements), all tiles 256B aligned by construction.
    char* ws = (char*)d_ws;
    const size_t bcn = (size_t)BATCH * CCH * NSP;       // 2M elems
    const size_t bnn = (size_t)BATCH * NSP * NSP;       // 32M elems
    __bf16* phi_p = (__bf16*)ws;                 ws += bcn * 2;            // 4 MiB
    __bf16* theta = (__bf16*)ws;                 ws += bcn * 2;            // 4 MiB
    __bf16* gmv_p = (__bf16*)ws;                 ws += bcn * 2;            // 4 MiB
    __bf16* wmkTp = (__bf16*)ws;                 ws += (size_t)NSP*NSP*2;  // 2 MiB
    __bf16* att1  = (__bf16*)ws;                 ws += bnn * 2;            // 64 MiB (also att3)
    float*  att2  = (float*)ws;                  ws += bnn * 4;            // 128 MiB
    float*  outb  = (float*)ws;                  /* 8 MiB */
    (void)in_sizes; (void)n_in; (void)out_size; (void)ws_size;

    // 1. w_mk -> bf16, transposed + K-panel packed
    wmk_transpose_kernel<<<dim3(NSP/32, NSP/32), 256, 0, stream>>>(w_mk, wmkTp);

    // 2. channel projections (phi / theta / g*w_mv^T) -> bf16 (B ops packed)
    proj_kernel<<<dim3(NSP/128, BATCH), 128, 0, stream>>>(
        x, w_phi, w_theta, w_g, w_mv, phi_p, theta, gmv_p);

    // 3. att1[b] = theta[b] (1024x64) @ phi[b] (64x1024)  -> bf16
    gemm_bf16_kernel<true><<<dim3(NSP/64, NSP/128, BATCH), 256, 0, stream>>>(
        theta, phi_p, att1,
        NSP, NSP, CCH, CCH, NSP,
        (long long)NSP * CCH, (long long)CCH * NSP, (long long)NSP * NSP);

    // 4. att2[b] = att1[b] (1024x1024) @ wmkT (1024x1024) -> f32  (dominant GEMM)
    gemm_bf16_kernel<false><<<dim3(NSP/64, NSP/128, BATCH), 256, 0, stream>>>(
        att1, wmkTp, att2,
        NSP, NSP, NSP, NSP, NSP,
        (long long)NSP * NSP, 0LL, (long long)NSP * NSP);

    // 5. column softmax (axis=1), write bf16 back into att1 buffer
    softmax_cols_kernel<<<dim3(NSP/256, BATCH), 256, 0, stream>>>(att2, att1);

    // 6. out[b] = att3[b] (1024x1024) @ gmv[b] (1024x64) -> f32 [b][n][c]
    gemm_bf16_kernel<false><<<dim3(CCH/64, NSP/128, BATCH), 256, 0, stream>>>(
        att1, gmv_p, outb,
        NSP, CCH, NSP, NSP, CCH,
        (long long)NSP * NSP, (long long)NSP * CCH, (long long)NSP * CCH);

    // 7. y = w_mask @ out^T + x
    mask_add_kernel<<<dim3(NSP/128, BATCH), 128, 0, stream>>>(outb, w_mask, x, y);
}
